// ContrastiveGNNLSTM_50766513439455
// MI455X (gfx1250) — matrix operand order repaired and checked by reading, not compiled, not run on hardware
//
#include <hip/hip_runtime.h>

typedef _Float16 v8h   __attribute__((ext_vector_type(8)));
typedef _Float16 v16h  __attribute__((ext_vector_type(16)));
typedef float    v8f   __attribute__((ext_vector_type(8)));
typedef float    f32x4 __attribute__((ext_vector_type(4)));

#define WMMA_F16(a, b, c) \
  __builtin_amdgcn_wmma_f32_16x16x32_f16(false, (a), false, (b), (short)0, (c), false, false)

#define LOG2E 1.4426950408889634f

__device__ __forceinline__ v16h cat8(v8h lo, v8h hi) {
  return __builtin_shufflevector(lo, hi, 0,1,2,3,4,5,6,7,8,9,10,11,12,13,14,15);
}
__device__ __forceinline__ v8f zero8() {
  v8f z;
#pragma unroll
  for (int i = 0; i < 8; ++i) z[i] = 0.0f;
  return z;
}
// Branchless hardware transcendentals: v_exp_f32 / v_rcp_f32 / v_log_f32.
__device__ __forceinline__ float fast_exp(float x) {
  return __builtin_amdgcn_exp2f(LOG2E * x);
}
__device__ __forceinline__ float fast_sigmoid(float x) {
  return __builtin_amdgcn_rcpf(1.0f + __builtin_amdgcn_exp2f(-LOG2E * x));
}
__device__ __forceinline__ float fast_tanh(float x) {
  return 1.0f - 2.0f * __builtin_amdgcn_rcpf(1.0f + __builtin_amdgcn_exp2f(2.0f * LOG2E * x));
}
__device__ __forceinline__ float fast_log(float x) {
  return 0.6931471805599453f * __builtin_amdgcn_logf(x);
}

// ---------------------------------------------------------------------------
// Kernel 1: LSTM encode. 12288 sequences, one wave per 16-seq tile.
// z[16,128] = A[16,64] * B[64,128] via 16x v_wmma_f32_16x16x32_f16 per step.
// Gate math split into two half-passes so only 4 f32 accumulators are live
// (keeps VGPR pressure < 256, avoids s_set_vgpr_msb churn).
// ---------------------------------------------------------------------------
__global__ __launch_bounds__(128) void lstm_kernel(
    const float* __restrict__ anchor, const float* __restrict__ pos,
    const float* __restrict__ neg, const float* __restrict__ Wih,
    const float* __restrict__ Whh, const float* __restrict__ blstm,
    float* __restrict__ hn) {
  __shared__ __attribute__((aligned(16))) _Float16 hsh[4][16 * 32];

  const int tid  = threadIdx.x;
  const int wv   = tid >> 5;
  const int lane = tid & 31;
  const int col  = lane & 15;
  const bool hi  = lane >= 16;
  const int w    = blockIdx.x * 4 + wv;   // global wave id: 0..767
  const int s0   = w * 16;                // first sequence of this tile

  // Map tile -> source tensor. All sources share inner layout [t][n][f].
  const float* src;
  int n0;
  if (s0 < 2048) {
    src = anchor + (s0 >> 7) * (64 * 128 * 16);
    n0  = s0 & 127;
  } else if (s0 < 4096) {
    int sp = s0 - 2048;
    src = pos + (sp >> 7) * (64 * 128 * 16);
    n0  = sp & 127;
  } else {
    int sp = s0 - 4096;
    src = neg + (sp >> 7) * (64 * 128 * 16);
    n0  = sp & 127;
  }

  // B fragments for W^T (K=48 real, padded to 64; N=128 gates -> 8 tiles x 2 K-chunks)
  v16h bf[8][2];
#pragma unroll
  for (int nt = 0; nt < 8; ++nt) {
#pragma unroll
    for (int kc = 0; kc < 2; ++kc) {
      const int n  = nt * 16 + col;
      const int ko = kc * 32 + (hi ? 8 : 0);
      v16h t;
#pragma unroll
      for (int hh = 0; hh < 16; ++hh) {
        const int k = ko + (hh < 8 ? hh : hh + 8);
        float v = 0.0f;
        if (k < 16)       v = Wih[n * 16 + k];
        else if (k < 48)  v = Whh[n * 32 + (k - 16)];
        t[hh] = (_Float16)v;
      }
      bf[nt][kc] = t;
    }
  }
  float bb[8];
#pragma unroll
  for (int nt = 0; nt < 8; ++nt) bb[nt] = blstm[nt * 16 + col];

  // init h (LDS, f16 [m][j]) and c (registers, C/D layout)
  _Float16* hrow = &hsh[wv][0];
  for (int i = lane; i < 512; i += 32) hrow[i] = (_Float16)0.0f;
  float creg[16];
#pragma unroll
  for (int i = 0; i < 16; ++i) creg[i] = 0.0f;

  v8h zz;
#pragma unroll
  for (int i = 0; i < 8; ++i) zz[i] = (_Float16)0.0f;

  for (int t = 0; t < 64; ++t) {
    // x features: 8 f32 per lane, convert to f16
    const float* xp = src + t * (128 * 16) + (n0 + col) * 16 + (hi ? 8 : 0);
    if (t < 63) __builtin_prefetch(xp + 128 * 16, 0, 1);
    f32x4 x0 = *(const f32x4*)xp;
    f32x4 x1 = *(const f32x4*)(xp + 4);
    v8h xh;
#pragma unroll
    for (int i = 0; i < 4; ++i) {
      xh[i]     = (_Float16)x0[i];
      xh[4 + i] = (_Float16)x1[i];
    }
    // h from LDS in A-fragment order
    v8h h0  = *(const v8h*)&hrow[col * 32 + (hi ? 8 : 0)];
    v8h h1v = *(const v8h*)&hrow[col * 32 + 16 + (hi ? 8 : 0)];
    v16h a0 = cat8(xh, h0);    // K 0..31  : features | h[0..15]
    v16h a1 = cat8(h1v, zz);   // K 32..63 : h[16..31] | zero pad

    // two half-passes: jt=0 -> gate tiles {0,2,4,6}, jt=1 -> {1,3,5,7}
#pragma unroll
    for (int jt = 0; jt < 2; ++jt) {
      v8f zi, zf, zg, zo;
#pragma unroll
      for (int i = 0; i < 8; ++i) {
        zi[i] = bb[0 + jt];
        zf[i] = bb[2 + jt];
        zg[i] = bb[4 + jt];
        zo[i] = bb[6 + jt];
      }
      zi = WMMA_F16(a0, bf[0 + jt][0], zi);
      zf = WMMA_F16(a0, bf[2 + jt][0], zf);
      zg = WMMA_F16(a0, bf[4 + jt][0], zg);
      zo = WMMA_F16(a0, bf[6 + jt][0], zo);
      zi = WMMA_F16(a1, bf[0 + jt][1], zi);
      zf = WMMA_F16(a1, bf[2 + jt][1], zf);
      zg = WMMA_F16(a1, bf[4 + jt][1], zg);
      zo = WMMA_F16(a1, bf[6 + jt][1], zo);
#pragma unroll
      for (int v = 0; v < 8; ++v) {
        const float iv = fast_sigmoid(zi[v]);
        const float fv = fast_sigmoid(zf[v]);
        const float gv = fast_tanh(zg[v]);
        const float ov = fast_sigmoid(zo[v]);
        float c = fv * creg[jt * 8 + v] + iv * gv;
        creg[jt * 8 + v] = c;
        const float hv = ov * fast_tanh(c);
        const int m = hi ? (v + 8) : v;
        hrow[m * 32 + jt * 16 + col] = (_Float16)hv;
      }
    }
  }

  // final hidden state -> workspace [s][32] f32
  for (int i = lane; i < 512; i += 32) hn[s0 * 32 + i] = (float)hrow[i];
}

// ---------------------------------------------------------------------------
// Kernel 2/3: GATv2 layer (Hin = O = 32). One block per (graph, head).
// fs/fd + aggregation via WMMA; logits (LeakyReLU inside sum) via VALU with
// online softmax; attn cached as f16 in LDS. h16s is reused to hold fs in
// row-major f16 so the logits loop reads 16B LDS broadcasts.
// ---------------------------------------------------------------------------
__global__ __launch_bounds__(128) void gat_kernel(
    const float* __restrict__ hin,   // [G][128][32]
    const float* __restrict__ Ws,    // [nh][32][32]
    const float* __restrict__ Wd,    // [nh][32][32]
    const float* __restrict__ avec,  // [nh][32]
    const float* __restrict__ bias,  // [nh][32]
    float* __restrict__ hout,        // [G][128][32]
    int nheads, float scale, int accumulate) {
  __shared__ __attribute__((aligned(16))) _Float16 h16s[128 * 32];   // h, later fs row-major
  __shared__ __attribute__((aligned(16))) _Float16 fs16t[32 * 128];  // [k][node]
  __shared__ __attribute__((aligned(16))) _Float16 fd16t[32 * 128];  // [k][node]
  __shared__ __attribute__((aligned(16))) _Float16 attn[128 * 136];  // padded rows

  const int tid  = threadIdx.x;
  const int g    = blockIdx.x / nheads;
  const int hd   = blockIdx.x % nheads;
  const int wv   = tid >> 5;
  const int lane = tid & 31;
  const int colL = lane & 15;
  const bool hi  = lane >= 16;

  // stage h_g as f16
  for (int i = tid; i < 4096; i += 128) h16s[i] = (_Float16)hin[g * 4096 + i];

  // weight B-fragments (K=32, N=32 -> 2 tiles each)
  const float* WsH = Ws + hd * 1024;
  const float* WdH = Wd + hd * 1024;
  v16h Bs[2], Bd[2];
#pragma unroll
  for (int nt = 0; nt < 2; ++nt) {
    const int n = nt * 16 + colL;
    v16h ts, td;
#pragma unroll
    for (int hh = 0; hh < 16; ++hh) {
      const int k = (hi ? 8 : 0) + (hh < 8 ? hh : hh + 8);
      ts[hh] = (_Float16)WsH[k * 32 + n];
      td[hh] = (_Float16)WdH[k * 32 + n];
    }
    Bs[nt] = ts;
    Bd[nt] = td;
  }
  __syncthreads();

  // fs = h@Ws, fd = h@Wd : 128x32x32, stored f16 transposed [k][node]
#pragma unroll
  for (int mi = 0; mi < 2; ++mi) {
    const int mt  = wv * 2 + mi;
    const int row = mt * 16 + colL;
    v8h alo = *(const v8h*)&h16s[row * 32 + (hi ? 8 : 0)];
    v8h ahi = *(const v8h*)&h16s[row * 32 + 16 + (hi ? 8 : 0)];
    v16h a  = cat8(alo, ahi);
#pragma unroll
    for (int nt = 0; nt < 2; ++nt) {
      v8f s_ = WMMA_F16(a, Bs[nt], zero8());
      v8f d_ = WMMA_F16(a, Bd[nt], zero8());
      const int n     = nt * 16 + colL;
      const int mbase = mt * 16 + (hi ? 8 : 0);
      v8h s8, d8;
#pragma unroll
      for (int v = 0; v < 8; ++v) {
        s8[v] = (_Float16)s_[v];
        d8[v] = (_Float16)d_[v];
      }
      *(v8h*)&fs16t[n * 128 + mbase] = s8;
      *(v8h*)&fd16t[n * 128 + mbase] = d8;
    }
  }
  __syncthreads();

  // h16s is dead now: fill it with fs row-major [node][k] for the logits loop
  for (int i = tid; i < 4096; i += 128) {
    h16s[(i >> 5) * 32 + (i & 31)] = fs16t[(i & 31) * 128 + (i >> 5)];
  }

  // per-thread gather of fd row + attention vector
  float fdv[32], avv[32];
#pragma unroll
  for (int k = 0; k < 32; ++k) {
    fdv[k] = (float)fd16t[k * 128 + tid];
    avv[k] = avec[hd * 32 + k];
  }
  __syncthreads();

  // logits row d = tid; online max/sum; cache raw e as f16
  float mrun = -3.0e38f, lrun = 0.0f;
  for (int s = 0; s < 128; ++s) {
    float e;
    if (s == tid) {
      e = -1.0e4f;  // diagonal mask (-inf)
    } else {
      e = 0.0f;
#pragma unroll
      for (int q = 0; q < 4; ++q) {
        v8h fsv = *(const v8h*)&h16s[s * 32 + q * 8];
#pragma unroll
        for (int j = 0; j < 8; ++j) {
          const float x  = fdv[q * 8 + j] + (float)fsv[j];
          const float lr = fmaxf(x, 0.0f) + 0.2f * fminf(x, 0.0f);
          e = fmaf(lr, avv[q * 8 + j], e);
        }
      }
    }
    attn[tid * 136 + s] = (_Float16)e;
    const float mn = fmaxf(mrun, e);
    lrun = lrun * fast_exp(mrun - mn) + fast_exp(e - mn);
    mrun = mn;
  }
  const float inv = __builtin_amdgcn_rcpf(lrun);
  for (int s = 0; s < 128; ++s) {
    const float e = (float)attn[tid * 136 + s];
    attn[tid * 136 + s] = (_Float16)(fast_exp(e - mrun) * inv);
  }
  __syncthreads();

  // out = attn[128x128] @ fs[128x32] : WMMA, K tiles = 4
  v8f oacc[2][2];
#pragma unroll
  for (int mi = 0; mi < 2; ++mi)
#pragma unroll
    for (int nt = 0; nt < 2; ++nt) oacc[mi][nt] = zero8();

  for (int kc = 0; kc < 4; ++kc) {
#pragma unroll
    for (int mi = 0; mi < 2; ++mi) {
      const int row = (wv * 2 + mi) * 16 + colL;
      v8h alo = *(const v8h*)&attn[row * 136 + kc * 32 + (hi ? 8 : 0)];
      v8h ahi = *(const v8h*)&attn[row * 136 + kc * 32 + 16 + (hi ? 8 : 0)];
      v16h a  = cat8(alo, ahi);
#pragma unroll
      for (int nt = 0; nt < 2; ++nt) {
        const int kb = (nt * 16 + colL) * 128 + kc * 32 + (hi ? 8 : 0);
        v8h blo = *(const v8h*)&fs16t[kb];
        v8h bhi = *(const v8h*)&fs16t[kb + 16];
        oacc[mi][nt] = WMMA_F16(a, cat8(blo, bhi), oacc[mi][nt]);
      }
    }
  }

  // + bias, write (or head-mean accumulate)
#pragma unroll
  for (int mi = 0; mi < 2; ++mi) {
#pragma unroll
    for (int nt = 0; nt < 2; ++nt) {
      const int n    = nt * 16 + colL;
      const float bv = bias[hd * 32 + n];
#pragma unroll
      for (int v = 0; v < 8; ++v) {
        const int m     = (wv * 2 + mi) * 16 + (hi ? (v + 8) : v);
        const float val = (oacc[mi][nt][v] + bv) * scale;
        float* p = &hout[(g * 128 + m) * 32 + n];
        if (accumulate) unsafeAtomicAdd(p, val);
        else            *p = val;
      }
    }
  }
}

// ---------------------------------------------------------------------------
__global__ void zero_kernel(float* __restrict__ p, int n) {
  const int i = blockIdx.x * blockDim.x + threadIdx.x;
  if (i < n) p[i] = 0.0f;
}

__global__ void pool_kernel(const float* __restrict__ h2, float* __restrict__ pooled) {
  const int g = blockIdx.x;
  const int k = threadIdx.x;  // 32 threads
  float s = 0.0f;
  for (int n = 0; n < 128; ++n) s += h2[(g * 128 + n) * 32 + k];
  pooled[g * 32 + k] = s * (1.0f / 128.0f);
}

__global__ void final_kernel(const float* __restrict__ pooled,
                             const float* __restrict__ hideout,
                             const float* __restrict__ timestep,
                             float* __restrict__ out) {
  __shared__ float lsh[16];
  const int b = threadIdx.x;  // 32 threads, first 16 active
  if (b < 16) {
    const float* an = pooled + b * 32;
    const float* po = pooled + (16 + b) * 32;
    float na = 0.0f, np = 0.0f;
    for (int k = 0; k < 32; ++k) { na += an[k] * an[k]; np += po[k] * po[k]; }
    na = fmaxf(sqrtf(na), 1e-6f);
    np = fmaxf(sqrtf(np), 1e-6f);
    float sims[5];
    float sp = 0.0f;
    for (int k = 0; k < 32; ++k) sp += (an[k] / na) * (po[k] / np);
    sims[4] = sp;
    for (int kk = 0; kk < 4; ++kk) {
      const float* ng = pooled + (32 + b * 4 + kk) * 32;
      float nn = 0.0f, dp = 0.0f;
      for (int k = 0; k < 32; ++k) nn += ng[k] * ng[k];
      nn = fmaxf(sqrtf(nn), 1e-6f);
      for (int k = 0; k < 32; ++k) dp += (an[k] / na) * (ng[k] / nn);
      sims[kk] = dp;
    }
    float m = sims[0];
    for (int i = 1; i < 5; ++i) m = fmaxf(m, sims[i]);
    float se = 0.0f;
    for (int i = 0; i < 5; ++i) se += fast_exp(sims[i] - m);
    lsh[b] = -(sims[4] - m - fast_log(se));
    // res = concat(anchor_res, hideout, timestep)
    for (int k = 0; k < 32; ++k) out[b * 35 + k] = an[k];
    out[b * 35 + 32] = hideout[b * 2 + 0];
    out[b * 35 + 33] = hideout[b * 2 + 1];
    out[b * 35 + 34] = timestep[b];
  }
  __syncthreads();
  if (b == 0) {
    float s = 0.0f;
    for (int i = 0; i < 16; ++i) s += lsh[i];
    out[560] = s * (1.0f / 16.0f);
  }
}

// ---------------------------------------------------------------------------
extern "C" void kernel_launch(void* const* d_in, const int* in_sizes, int n_in,
                              void* d_out, int out_size, void* d_ws, size_t ws_size,
                              hipStream_t stream) {
  (void)in_sizes; (void)n_in; (void)out_size; (void)ws_size;
  const float* anchor   = (const float*)d_in[0];
  const float* pos      = (const float*)d_in[1];
  const float* neg      = (const float*)d_in[2];
  const float* hideout  = (const float*)d_in[3];
  const float* timestep = (const float*)d_in[4];
  const float* Wih      = (const float*)d_in[5];
  const float* Whh      = (const float*)d_in[6];
  const float* blstm    = (const float*)d_in[7];
  const float* W1s      = (const float*)d_in[8];
  const float* W1d      = (const float*)d_in[9];
  const float* a1       = (const float*)d_in[10];
  const float* b1       = (const float*)d_in[11];
  const float* W2s      = (const float*)d_in[12];
  const float* W2d      = (const float*)d_in[13];
  const float* a2       = (const float*)d_in[14];
  const float* b2       = (const float*)d_in[15];

  float* ws     = (float*)d_ws;
  float* hn     = ws;                // 12288*32 = 393216
  float* h1     = ws + 393216;       // 96*128*32
  float* h2     = ws + 786432;       // 96*128*32
  float* pooled = ws + 1179648;      // 96*32

  lstm_kernel<<<192, 128, 0, stream>>>(anchor, pos, neg, Wih, Whh, blstm, hn);
  zero_kernel<<<(393216 + 255) / 256, 256, 0, stream>>>(h1, 393216);
  gat_kernel<<<96 * 8, 128, 0, stream>>>(hn, W1s, W1d, a1, b1, h1, 8, 0.125f, 1);
  gat_kernel<<<96, 128, 0, stream>>>(h1, W2s, W2d, a2, b2, h2, 1, 1.0f, 0);
  pool_kernel<<<96, 32, 0, stream>>>(h2, pooled);
  final_kernel<<<1, 32, 0, stream>>>(pooled, hideout, timestep, (float*)d_out);
}
